// MultiHeadAttention_34720515621293
// MI455X (gfx1250) — matrix operand order repaired and checked by reading, not compiled, not run on hardware
//
#include <hip/hip_runtime.h>
#include <hip/hip_bf16.h>

// ---------------- types for WMMA fragments ----------------
typedef _Float16 h16;
typedef __attribute__((ext_vector_type(16))) _Float16 v16h;
typedef __attribute__((ext_vector_type(8)))  _Float16 v8h;
typedef __attribute__((ext_vector_type(8)))  float    v8f;

union V16 { v16h v; v8h h[2]; };

__device__ __forceinline__ v8f wmma_f16(v16h a, v16h b, v8f c) {
  // D = A(16x32 f16) x B(32x16 f16) + C(16x16 f32)
  return __builtin_amdgcn_wmma_f32_16x16x32_f16(
      /*neg_a=*/false, a, /*neg_b=*/false, b,
      /*c_mod=*/(short)0, c, /*reuse_a=*/false, /*reuse_b=*/false);
}

#define D_MODEL 1024
#define HEADS   16
#define DK      64
#define BS      2
#define SEQ     2048
#define MTOK    (BS * SEQ)          // 4096 tokens

// ---------------- fp32 -> f16 conversion ----------------
__global__ void cvt_f32_f16(const float* __restrict__ s, h16* __restrict__ d, int n) {
  int i = blockIdx.x * blockDim.x + threadIdx.x;
  int st = gridDim.x * blockDim.x;
  for (; i < n; i += st) d[i] = (h16)s[i];
}

// ---------------- WMMA GEMM: Y = X @ W^T + bias ----------------
// X: [MTOK, D_MODEL] f16 row-major, W: [D_MODEL(out), D_MODEL(in)] f16 row-major
// One wave computes a 16x64 output strip (4 N-tiles), K-loop in steps of 32.
// mode 0: write f16 head-split  [b,h,s,dk]
// mode 1: write f16 transposed  [b,h,dk,s]   (for V)
// mode 2: write f32 row-major   [MTOK, D_MODEL] (final output)
__global__ __launch_bounds__(128) void gemm16x64_wmma(
    const h16* __restrict__ A, const h16* __restrict__ W,
    const float* __restrict__ bias,
    h16* __restrict__ outH, float* __restrict__ outF, int mode)
{
  const int lane  = threadIdx.x & 31;
  const int wave  = threadIdx.x >> 5;
  const int wg    = blockIdx.x * 4 + wave;      // 4096 waves total
  const int mTile = wg >> 4;                    // 256 M-tiles
  const int nBlk  = wg & 15;                    // 16 N-blocks of 64
  const int m0 = mTile * 16, n0 = nBlk * 64;
  const int r  = lane & 15,  hi = lane >> 4;

  v8f acc[4] = {};

  const h16* aRow = A + (size_t)(m0 + r) * D_MODEL;

  for (int k0 = 0; k0 < D_MODEL; k0 += 32) {
    __builtin_prefetch(aRow + k0 + 128, 0, 0);
    V16 a;
    a.h[0] = *(const v8h*)(aRow + k0 + 8 * hi);
    a.h[1] = *(const v8h*)(aRow + k0 + 16 + 8 * hi);
#pragma unroll
    for (int t = 0; t < 4; ++t) {
      // B[k,n] = W[n0+t*16+n, k]; lane holds two 8-half segments at k 8*hi and 16+8*hi
      const h16* wp = W + (size_t)(n0 + t * 16 + r) * D_MODEL + k0 + 8 * hi;
      V16 b;
      b.h[0] = *(const v8h*)(wp);
      b.h[1] = *(const v8h*)(wp + 16);
      acc[t] = wmma_f16(a.v, b.v, acc[t]);
    }
  }

#pragma unroll
  for (int t = 0; t < 4; ++t) {
    const int n   = n0 + t * 16 + r;   // same column for all 8 elements
    const float bv_ = bias[n];
#pragma unroll
    for (int e = 0; e < 8; ++e) {
      const int m = m0 + e + 8 * hi;
      const float val = acc[t][e] + bv_;
      if (mode == 2) {
        outF[(size_t)m * D_MODEL + n] = val;
      } else {
        const int b  = m >> 11, s = m & (SEQ - 1);
        const int hh = n >> 6,  d = n & (DK - 1);
        if (mode == 0)
          outH[(((size_t)b * HEADS + hh) * SEQ + s) * DK + d] = (h16)val;
        else
          outH[(((size_t)b * HEADS + hh) * DK + d) * SEQ + s] = (h16)val;
      }
    }
  }
}

// ---------------- Flash attention (one wave per (b*h, 16-row q tile)) ----------------
// Qh,Kh: [BS*HEADS, SEQ, DK] f16 ; Vt: [BS*HEADS, DK, SEQ] f16
// scores: [BS*HEADS, SEQ, DK] f32 (output #2) ; concat: [BS, SEQ, D_MODEL] f16
__global__ __launch_bounds__(128) void attn_flash_wmma(
    const h16* __restrict__ Qh, const h16* __restrict__ Kh,
    const h16* __restrict__ Vt,
    float* __restrict__ scores, h16* __restrict__ concat)
{
  __shared__ h16 ldsP[4][16 * 32];   // per-wave P (16x32) staging for C->A transpose

  const int lane = threadIdx.x & 31;
  const int wave = threadIdx.x >> 5;
  const int wg   = blockIdx.x * 4 + wave;   // 4096 waves
  const int bh   = wg >> 7;                 // 32 (b*h) slabs
  const int qt   = wg & 127;                // 128 q-tiles
  const int q0   = qt * 16;
  const int r = lane & 15, hi = lane >> 4;

  const h16* Qb = Qh + (size_t)bh * SEQ * DK;
  const h16* Kb = Kh + (size_t)bh * SEQ * DK;
  const h16* Vb = Vt + (size_t)bh * DK * SEQ;

  // Q A-fragments: two K-steps over d (0..31, 32..63), kept in registers
  const h16* qrow = Qb + (size_t)(q0 + r) * DK;
  V16 qa0, qa1;
  qa0.h[0] = *(const v8h*)(qrow + 0  + 8 * hi);
  qa0.h[1] = *(const v8h*)(qrow + 16 + 8 * hi);
  qa1.h[0] = *(const v8h*)(qrow + 32 + 8 * hi);
  qa1.h[1] = *(const v8h*)(qrow + 48 + 8 * hi);

  float mrow[8], lrow[8];
#pragma unroll
  for (int e = 0; e < 8; ++e) { mrow[e] = -3.0e38f; lrow[e] = 0.0f; }
  v8f acc[4] = {};

  h16* myP = ldsP[wave];
  const float sm_scale = 0.125f;   // 1/sqrt(64)

  for (int j0 = 0; j0 < SEQ; j0 += 32) {
    __builtin_prefetch(Kb + (size_t)(j0 + 32 + r) * DK, 0, 0);

    // ---- S = Q @ K^T for 32 keys: two 16x16 N-tiles, 2 K-steps each ----
    const h16* krow0 = Kb + (size_t)(j0 + r) * DK;        // keys j0..j0+15
    const h16* krow1 = Kb + (size_t)(j0 + 16 + r) * DK;   // keys j0+16..j0+31
    V16 b00, b01, b10, b11;
    b00.h[0] = *(const v8h*)(krow0 + 0  + 8 * hi);
    b00.h[1] = *(const v8h*)(krow0 + 16 + 8 * hi);
    b01.h[0] = *(const v8h*)(krow0 + 32 + 8 * hi);
    b01.h[1] = *(const v8h*)(krow0 + 48 + 8 * hi);
    b10.h[0] = *(const v8h*)(krow1 + 0  + 8 * hi);
    b10.h[1] = *(const v8h*)(krow1 + 16 + 8 * hi);
    b11.h[0] = *(const v8h*)(krow1 + 32 + 8 * hi);
    b11.h[1] = *(const v8h*)(krow1 + 48 + 8 * hi);

    v8f s0 = {}, s1 = {};
    s0 = wmma_f16(qa0.v, b00.v, s0);
    s0 = wmma_f16(qa1.v, b01.v, s0);
    s1 = wmma_f16(qa0.v, b10.v, s1);
    s1 = wmma_f16(qa1.v, b11.v, s1);

    // ---- online softmax update (row m = e + 8*hi lives across 16 lanes) ----
    float alpha[8];
#pragma unroll
    for (int e = 0; e < 8; ++e) {
      float a0 = s0[e] * sm_scale;
      float a1 = s1[e] * sm_scale;
      float mx = fmaxf(a0, a1);
      mx = fmaxf(mx, __shfl_xor(mx, 1, 32));
      mx = fmaxf(mx, __shfl_xor(mx, 2, 32));
      mx = fmaxf(mx, __shfl_xor(mx, 4, 32));
      mx = fmaxf(mx, __shfl_xor(mx, 8, 32));
      const float mnew = fmaxf(mrow[e], mx);
      alpha[e] = __expf(mrow[e] - mnew);
      mrow[e]  = mnew;
      a0 = __expf(a0 - mnew);
      a1 = __expf(a1 - mnew);
      s0[e] = a0; s1[e] = a1;
      float rs = a0 + a1;
      rs += __shfl_xor(rs, 1, 32);
      rs += __shfl_xor(rs, 2, 32);
      rs += __shfl_xor(rs, 4, 32);
      rs += __shfl_xor(rs, 8, 32);
      lrow[e] = lrow[e] * alpha[e] + rs;
    }
#pragma unroll
    for (int t = 0; t < 4; ++t)
#pragma unroll
      for (int e = 0; e < 8; ++e) acc[t][e] *= alpha[e];

    // ---- transpose P (C-layout -> A-layout) through per-wave LDS tile ----
#pragma unroll
    for (int e = 0; e < 8; ++e) {
      const int row = e + 8 * hi;
      myP[row * 32 + r]      = (h16)s0[e];
      myP[row * 32 + 16 + r] = (h16)s1[e];
    }
    // same-wave DS ops are in-order; aliasing through myP orders the reads
    V16 pa;
    pa.h[0] = *(const v8h*)(myP + r * 32 + 8 * hi);
    pa.h[1] = *(const v8h*)(myP + r * 32 + 16 + 8 * hi);

    // ---- O += P @ V_block : 4 N-tiles over d, one K-step of 32 ----
#pragma unroll
    for (int t = 0; t < 4; ++t) {
      const h16* vrow = Vb + (size_t)(t * 16 + r) * SEQ + j0 + 8 * hi;
      V16 vb;
      vb.h[0] = *(const v8h*)(vrow);
      vb.h[1] = *(const v8h*)(vrow + 16);
      acc[t] = wmma_f16(pa.v, vb.v, acc[t]);
    }
  }

  // ---- finalize: O /= l ; write scores (f32) and concat (f16) ----
  const int b = bh >> 4, hh = bh & 15;
#pragma unroll
  for (int t = 0; t < 4; ++t) {
    const int d = t * 16 + r;
#pragma unroll
    for (int e = 0; e < 8; ++e) {
      const int m = q0 + e + 8 * hi;
      const float val = acc[t][e] / lrow[e];
      scores[((size_t)bh * SEQ + m) * DK + d] = val;
      concat[((size_t)b * SEQ + m) * D_MODEL + hh * DK + d] = (h16)val;
    }
  }
}

// ---------------- host-side orchestration ----------------
extern "C" void kernel_launch(void* const* d_in, const int* in_sizes, int n_in,
                              void* d_out, int out_size, void* d_ws, size_t ws_size,
                              hipStream_t stream) {
  (void)in_sizes; (void)n_in; (void)out_size; (void)ws_size;

  const float* q  = (const float*)d_in[0];
  const float* k  = (const float*)d_in[1];
  const float* v  = (const float*)d_in[2];
  const float* Wq = (const float*)d_in[3];
  const float* bq = (const float*)d_in[4];
  const float* Wk = (const float*)d_in[5];
  const float* bk = (const float*)d_in[6];
  const float* Wv = (const float*)d_in[7];
  const float* bv = (const float*)d_in[8];
  const float* Wo = (const float*)d_in[9];
  const float* bo = (const float*)d_in[10];

  float* out    = (float*)d_out;                       // [BS,SEQ,D_MODEL]
  float* scores = out + (size_t)BS * SEQ * D_MODEL;    // [BS,HEADS,SEQ,DK]

  // workspace carve-up (f16 elements): 64 MiB total
  const size_t NX = (size_t)MTOK * D_MODEL;            // 4 Mi elements
  const size_t NW = (size_t)D_MODEL * D_MODEL;         // 1 Mi elements
  h16* ws = (h16*)d_ws;
  h16* xq = ws;            h16* xk = xq + NX;  h16* xv = xk + NX;
  h16* wq = xv + NX;       h16* wk = wq + NW;  h16* wv = wk + NW;  h16* wo = wv + NW;
  h16* Qh = wo + NW;       h16* Kh = Qh + NX;  h16* Vt = Kh + NX;
  h16* cc = Vt + NX;

  // 1) fp32 -> f16 staging
  cvt_f32_f16<<<2048, 256, 0, stream>>>(q,  xq, (int)NX);
  cvt_f32_f16<<<2048, 256, 0, stream>>>(k,  xk, (int)NX);
  cvt_f32_f16<<<2048, 256, 0, stream>>>(v,  xv, (int)NX);
  cvt_f32_f16<<<1024, 256, 0, stream>>>(Wq, wq, (int)NW);
  cvt_f32_f16<<<1024, 256, 0, stream>>>(Wk, wk, (int)NW);
  cvt_f32_f16<<<1024, 256, 0, stream>>>(Wv, wv, (int)NW);
  cvt_f32_f16<<<1024, 256, 0, stream>>>(Wo, wo, (int)NW);

  // 2) projections: Q/K head-split, V transposed head-split
  gemm16x64_wmma<<<1024, 128, 0, stream>>>(xq, wq, bq, Qh, nullptr, 0);
  gemm16x64_wmma<<<1024, 128, 0, stream>>>(xk, wk, bk, Kh, nullptr, 0);
  gemm16x64_wmma<<<1024, 128, 0, stream>>>(xv, wv, bv, Vt, nullptr, 1);

  // 3) flash attention -> scores (f32 output #2) + concat (f16)
  attn_flash_wmma<<<1024, 128, 0, stream>>>(Qh, Kh, Vt, scores, cc);

  // 4) output projection -> f32 output #1
  gemm16x64_wmma<<<1024, 128, 0, stream>>>(cc, wo, bo, nullptr, out, 2);
}
